// ScaledDotProduct_9036611191412
// MI455X (gfx1250) — matrix operand order repaired and checked by reading, not compiled, not run on hardware
//
#include <hip/hip_runtime.h>

typedef __attribute__((ext_vector_type(16))) _Float16 v16h;
typedef __attribute__((ext_vector_type(8)))  _Float16 v8h;
typedef __attribute__((ext_vector_type(2)))  __fp16   v2hp;   // cvt_pkrtz result type
typedef __attribute__((ext_vector_type(8)))  float    v8f;

#define B_  2
#define H_  16
#define S_  2048
#define DK_ 64
#define CH_ 64               // keys per chunk
#define NCHUNK (S_ / CH_)

union U16h { v16h v; struct { v8h lo, hi; } s; };

__global__ __launch_bounds__(128) void
fa_masked_kernel(const float* __restrict__ Q, const float* __restrict__ K,
                 const float* __restrict__ V, const float* __restrict__ Mk,
                 float* __restrict__ O)
{
    // single-buffer K/V staging + per-wave P slab : 8+8+8 = 24 KB
    __shared__ __align__(64) _Float16 ksh[CH_ * 64];     // [key][dim]
    __shared__ __align__(64) _Float16 vtsh[64 * CH_];    // [dim][key]
    __shared__ __align__(64) _Float16 psh[4][16 * CH_];  // [row][key]

    const int tid  = threadIdx.x;
    const int wave = tid >> 5;
    const int lane = tid & 31;
    const int n    = lane & 15;
    const int hi   = lane >> 4;
    const int b    = blockIdx.z;
    const int h    = blockIdx.y;
    const int qbase = blockIdx.x * 64 + wave * 16;
    const size_t bh = (size_t)(b * H_ + h) * S_;
    const int kh = hi * 16;           // lane's 16 contiguous K-halves in B-frags

    // loop-invariant per-thread offsets (keep loads in saddr + voffset form)
    const int koff0 = (tid >> 4) * DK_ + (tid & 15) * 4;        // K staging
    const int voff0 = (tid >> 4) * 2 * DK_ + (tid & 15) * 4;    // V staging (key pairs)
    const int d4    = (tid & 15) * 4;
    const int key2  = (tid >> 4) * 2;
    const int moff0 = 8 * hi * S_ + n;                          // mask gather
    const int poff  = (tid >> 1) * DK_ + (tid & 1) * 32;        // prefetch lines

    // ---- Q A-fragments with 1/sqrt(64) * log2(e) folded in ----
    // (scores then feed v_exp_f32 (= exp2) directly; score sign preserved)
    const float QSCALE = 0.125f * 1.4426950408889634f;
    v16h qa0, qa1;
    {
        const float* qrow = Q + (bh + qbase + n) * DK_;
        #pragma unroll
        for (int v = 0; v < 8; ++v) {
            const int kk = hi * 16 + 2 * v;
            float2 f0 = *(const float2*)(qrow + kk);
            float2 f1 = *(const float2*)(qrow + 32 + kk);
            qa0[2 * v]     = (_Float16)(f0.x * QSCALE);
            qa0[2 * v + 1] = (_Float16)(f0.y * QSCALE);
            qa1[2 * v]     = (_Float16)(f1.x * QSCALE);
            qa1[2 * v + 1] = (_Float16)(f1.y * QSCALE);
        }
    }

    // all-ones B fragment: lacc = P x ones -> per-lane row-sum (no shuffles)
    v16h ones;
    #pragma unroll
    for (int i = 0; i < 16; ++i) ones[i] = (_Float16)1.0f;

    v8f acc[4];
    v8f lacc = (v8f){};
    #pragma unroll
    for (int t = 0; t < 4; ++t) acc[t] = (v8f){};

    for (int c = 0; c < NCHUNK; ++c) {
        const int kc = c * CH_;
        // uniform (SGPR) chunk base pointers; advance is scalar work
        const float* Kc = K + (bh + kc) * (size_t)DK_;
        const float* Vc = V + (bh + kc) * (size_t)DK_;
        const float* Mc = Mk + ((size_t)b * S_ + qbase) * S_ + kc;

        __syncthreads();   // previous chunk's readers done before restaging

        // ---- Stage K chunk: 64x64 f32 -> packed f16 LDS [key][dim] ----
        #pragma unroll
        for (int j = 0; j < 8; ++j) {
            float4 f = *(const float4*)(Kc + koff0 + j * (8 * DK_));
            *(v2hp*)(ksh + tid * 4 + j * 512)     = __builtin_amdgcn_cvt_pkrtz(f.x, f.y);
            *(v2hp*)(ksh + tid * 4 + j * 512 + 2) = __builtin_amdgcn_cvt_pkrtz(f.z, f.w);
        }
        // ---- Stage V chunk transposed (pairs of keys -> packed b32 stores) ----
        #pragma unroll
        for (int j = 0; j < 4; ++j) {
            float4 a = *(const float4*)(Vc + voff0 + j * (16 * DK_));
            float4 d = *(const float4*)(Vc + voff0 + j * (16 * DK_) + DK_);
            _Float16* vb = vtsh + d4 * CH_ + key2 + j * 16;
            *(v2hp*)(vb + 0 * CH_) = __builtin_amdgcn_cvt_pkrtz(a.x, d.x);
            *(v2hp*)(vb + 1 * CH_) = __builtin_amdgcn_cvt_pkrtz(a.y, d.y);
            *(v2hp*)(vb + 2 * CH_) = __builtin_amdgcn_cvt_pkrtz(a.z, d.z);
            *(v2hp*)(vb + 3 * CH_) = __builtin_amdgcn_cvt_pkrtz(a.w, d.w);
        }
        __syncthreads();

        // ---- register-free prefetch hints for next chunk (global_prefetch_b8) ----
        if (c + 1 < NCHUNK) {
            __builtin_prefetch(Kc + CH_ * DK_ + poff, 0, 3);
            __builtin_prefetch(Vc + CH_ * DK_ + poff, 0, 3);
        }

        // ---- QK^T: four 16x16 score tiles, K-dim split 0..31 / 32..63 ----
        v8f s[4];
        #pragma unroll
        for (int t = 0; t < 4; ++t) {
            s[t] = (v8f){};
            s[t] = __builtin_amdgcn_wmma_f32_16x16x32_f16(false, qa0, false,
                       *(const v16h*)(ksh + (t * 16 + n) * 64 + kh),
                       (short)0, s[t], false, false);
            s[t] = __builtin_amdgcn_wmma_f32_16x16x32_f16(false, qa1, false,
                       *(const v16h*)(ksh + (t * 16 + n) * 64 + 32 + kh),
                       (short)0, s[t], false, false);
        }

        // ---- mask * positive-threshold * exp2 (no max subtraction needed:
        //      kept scores bounded, exp stays far inside f32/f16 range) ----
        _Float16* pp = psh[wave];
        #pragma unroll
        for (int v = 0; v < 8; ++v) {
            const int r = v + 8 * hi;
            #pragma unroll
            for (int t = 0; t < 4; ++t) {
                float a  = s[t][v] * Mc[moff0 + v * S_ + t * 16];
                float ax = (a > 0.0f) ? a : -10000.0f;       // exp2 -> 0
                pp[r * CH_ + t * 16 + n] = (_Float16)__builtin_amdgcn_exp2f(ax);
            }
        }

        __builtin_amdgcn_wave_barrier();   // order per-wave LDS store -> load

        // ---- P A-fragments via direct b128 loads (A subtile = 16 contiguous
        //      K halves per lane: base + blk*16) ----
        U16h pa0, pa1;
        const _Float16* ppn = pp + n * CH_ + hi * 16;
        pa0.s.lo = *(const v8h*)(ppn);
        pa0.s.hi = *(const v8h*)(ppn + 8);
        pa1.s.lo = *(const v8h*)(ppn + 32);
        pa1.s.hi = *(const v8h*)(ppn + 40);

        // ---- O += P*V ; l += P*ones ----
        #pragma unroll
        for (int nt = 0; nt < 4; ++nt) {
            acc[nt] = __builtin_amdgcn_wmma_f32_16x16x32_f16(false, pa0.v, false,
                          *(const v16h*)(vtsh + (nt * 16 + n) * CH_ + kh),
                          (short)0, acc[nt], false, false);
            acc[nt] = __builtin_amdgcn_wmma_f32_16x16x32_f16(false, pa1.v, false,
                          *(const v16h*)(vtsh + (nt * 16 + n) * CH_ + 32 + kh),
                          (short)0, acc[nt], false, false);
        }
        lacc = __builtin_amdgcn_wmma_f32_16x16x32_f16(false, pa0.v, false, ones,
                   (short)0, lacc, false, false);
        lacc = __builtin_amdgcn_wmma_f32_16x16x32_f16(false, pa1.v, false, ones,
                   (short)0, lacc, false, false);
    }

    // ---- epilogue: normalize and store (coalesced f32; saddr + voffset) ----
    float* Ob = O + (bh + qbase) * DK_;      // uniform base
    #pragma unroll
    for (int v = 0; v < 8; ++v) {
        const float inv = __builtin_amdgcn_rcpf(lacc[v]);
        const int obase = (v + 8 * hi) * DK_ + n;    // thread part
        #pragma unroll
        for (int nt = 0; nt < 4; ++nt)
            Ob[obase + nt * 16] = acc[nt][v] * inv;
    }
}

extern "C" void kernel_launch(void* const* d_in, const int* in_sizes, int n_in,
                              void* d_out, int out_size, void* d_ws, size_t ws_size,
                              hipStream_t stream) {
    (void)in_sizes; (void)n_in; (void)out_size; (void)d_ws; (void)ws_size;
    const float* Q  = (const float*)d_in[0];
    const float* K  = (const float*)d_in[1];
    const float* V  = (const float*)d_in[2];
    const float* Mk = (const float*)d_in[3];
    float* O = (float*)d_out;

    dim3 grid(S_ / 64, H_, B_);   // 32 q-tiles x 16 heads x 2 batches = 1024 blocks
    dim3 block(128);              // 4 wave32, one 16-row Q tile per wave
    fa_masked_kernel<<<grid, block, 0, stream>>>(Q, K, V, Mk, O);
}
